// BarebonesConventionalAttention_51084341018885
// MI455X (gfx1250) — compile-verified, matrix operand order checked by reading, not run on hardware
//
#include <hip/hip_runtime.h>

// ---------------------------------------------------------------------------
// Causal flash attention, B=2,H=16,S=2048,D=64, fp32 in/out, gfx1250.
// Transposed formulation: S^T = K x Q^T and O^T = V^T x P^T so the softmax
// key-reduction stays in registers (one shfl_xor(16) per reduction). f16 WMMA
// 16x16x32 with f32 accumulation; online softmax in the exp2 domain (log2e
// folded into the Q scale so v_exp_f32 is used raw, no per-element mul).
// 8 wave32 waves per block, each owns 16 query columns.
// ---------------------------------------------------------------------------

typedef _Float16 v16h __attribute__((ext_vector_type(16)));
typedef _Float16 v8h  __attribute__((ext_vector_type(8)));
typedef _Float16 v4h  __attribute__((ext_vector_type(4)));
typedef float    v8f  __attribute__((ext_vector_type(8)));

#define S_SEQ   2048
#define D_HEAD  64
#define N_HEADS 16
#define N_BATCH 2
#define QBLK    128   // query rows per workgroup (8 waves x 16)
#define KBLK    64    // key rows per LDS tile
#define LDSS    72    // LDS row stride in halves: 144B, 16B-aligned, bank-staggered

__device__ __forceinline__ float fast_exp2(float x) {
#if __has_builtin(__builtin_amdgcn_exp2f)
  return __builtin_amdgcn_exp2f(x);   // raw v_exp_f32
#else
  return exp2f(x);
#endif
}

// 16x32 f16 A/B fragment (CDNA5 K-packing):
//   lanes 0-15 : K = {k0..k0+7} then {k0+16..k0+23}
//   lanes 16-31: K = {k0+8..k0+15} then {k0+24..k0+31}
// caller passes p = rowBase + k0 + hi*8  (16B aligned)
__device__ __forceinline__ v16h load_frag16(const _Float16* p) {
  v8h lo = *(const v8h*)(p);
  v8h hi = *(const v8h*)(p + 16);
  v16h r;
#pragma unroll
  for (int i = 0; i < 8; ++i) { r[i] = lo[i]; r[i + 8] = hi[i]; }
  return r;
}

__device__ __forceinline__ v8f wmma_f16(v16h a, v16h b, v8f c) {
  return __builtin_amdgcn_wmma_f32_16x16x32_f16(false, a, false, b, (short)0, c,
                                                false, false);
}

__global__ __launch_bounds__(256)
void fa_causal_gfx1250(const float* __restrict__ Q, const float* __restrict__ K,
                       const float* __restrict__ V, float* __restrict__ O) {
  __shared__ __align__(16) _Float16 ldsK[KBLK * LDSS];    // K tile   [key][dim]
  __shared__ __align__(16) _Float16 ldsV[D_HEAD * LDSS];  // V^T tile [dim][key]
  __shared__ __align__(16) _Float16 ldsP[8 * 16 * LDSS];  // per-wave P^T [q][key]

  const int qb   = blockIdx.x;
  const int h    = blockIdx.y;
  const int b    = blockIdx.z;
  const int t    = threadIdx.x;
  const int wave = t >> 5;
  const int lane = t & 31;
  const int ln16 = lane & 15;
  const int hi   = lane >> 4;

  const size_t base = ((size_t)(b * N_HEADS + h)) * S_SEQ * D_HEAD;
  const float* Qb = Q + base;
  const float* Kb = K + base;
  const float* Vb = V + base;
  float*       Ob = O + base;

  const int qw = qb * QBLK + wave * 16;  // first q row of this wave's tile
  const int qg = qw + ln16;              // this lane's query row (column of S^T)

  // Q^T B-fragments, pre-scaled by (1/sqrt(D)) * log2(e) -> exp2-domain scores
  const float qscale = 0.125f * 1.44269504088896340736f;
  v16h qa[2];
#pragma unroll
  for (int kk = 0; kk < 2; ++kk) {
    const float* qp = Qb + (size_t)qg * D_HEAD + kk * 32 + hi * 8;
    v16h a;
#pragma unroll
    for (int i = 0; i < 8; ++i) a[i]     = (_Float16)(qp[i]      * qscale);
#pragma unroll
    for (int i = 0; i < 8; ++i) a[i + 8] = (_Float16)(qp[16 + i] * qscale);
    qa[kk] = a;
  }

  // O^T accumulators: o[j] rows = dims j*16 + r + 8*hi, col = this lane's q
  v8f o[4];
#pragma unroll
  for (int j = 0; j < 4; ++j)
#pragma unroll
    for (int r = 0; r < 8; ++r) o[j][r] = 0.f;
  float m = -1e30f, l = 0.f;   // per-lane scalars, exp2 domain

  const int kbMax = 2 * qb + 1;
  for (int kb = 0; kb <= kbMax; ++kb) {
    // ---- stage K row-major (f32->f16), coalesced b128 loads, b64 LDS stores
#pragma unroll
    for (int i = 0; i < 4; ++i) {
      const int g = t + i * 256;     // 0..1023 float4 groups
      const int r = g >> 4;          // key 0..63
      const int c = (g & 15) << 2;   // dim 0,4,..,60
      const float4 k4 = *(const float4*)(Kb + (size_t)(kb * KBLK + r) * D_HEAD + c);
      v4h kh = {(_Float16)k4.x, (_Float16)k4.y, (_Float16)k4.z, (_Float16)k4.w};
      *(v4h*)&ldsK[r * LDSS + c] = kh;
    }
    // ---- stage V transposed via per-thread 4x4 register transpose
    {
      const int rb = (t >> 4) << 2;  // key base
      const int cb = (t & 15) << 2;  // dim base (coalesced global)
      float4 vr[4];
#pragma unroll
      for (int i = 0; i < 4; ++i)
        vr[i] = *(const float4*)(Vb + (size_t)(kb * KBLK + rb + i) * D_HEAD + cb);
#pragma unroll
      for (int j = 0; j < 4; ++j) {
        v4h col = {(_Float16)(((const float*)&vr[0])[j]),
                   (_Float16)(((const float*)&vr[1])[j]),
                   (_Float16)(((const float*)&vr[2])[j]),
                   (_Float16)(((const float*)&vr[3])[j])};
        *(v4h*)&ldsV[(cb + j) * LDSS + rb] = col;
      }
    }
    // pull next tile toward L2 while computing on this one
    if (kb < kbMax) {
      const size_t nb = (size_t)(kb + 1) * KBLK * D_HEAD;
      if (t < 64)       __builtin_prefetch(Kb + nb + (size_t)t * D_HEAD, 0, 1);
      else if (t < 128) __builtin_prefetch(Vb + nb + (size_t)(t - 64) * D_HEAD, 0, 1);
    }
    __syncthreads();

    // ---- S^T = K x Q^T: s[j] rows = keys j*16 + r + 8*hi, col = lane's q
    v8f s[4];
#pragma unroll
    for (int j = 0; j < 4; ++j) {
      const _Float16* kr = &ldsK[(j * 16 + ln16) * LDSS + hi * 8];
      const v16h ka0 = load_frag16(kr);
      const v16h ka1 = load_frag16(kr + 32);
      v8f c;
#pragma unroll
      for (int r = 0; r < 8; ++r) c[r] = 0.f;
      c = wmma_f16(ka0, qa[0], c);
      c = wmma_f16(ka1, qa[1], c);
      s[j] = c;
    }

    // ---- causal mask: key = kb*64 + j*16 + r + 8*hi, query = qg (per lane)
    if (kb * KBLK + (KBLK - 1) > qw) {
      const int kbase = kb * KBLK + 8 * hi;
#pragma unroll
      for (int j = 0; j < 4; ++j)
#pragma unroll
        for (int r = 0; r < 8; ++r)
          if (kbase + j * 16 + r > qg) s[j][r] = -1e30f;
    }

    // ---- row max: pairwise tree (depth 5) + one xor-16 half-merge
    v8f t01, t23, tm;
#pragma unroll
    for (int r = 0; r < 8; ++r) {
      t01[r] = fmaxf(s[0][r], s[1][r]);
      t23[r] = fmaxf(s[2][r], s[3][r]);
      tm[r]  = fmaxf(t01[r], t23[r]);
    }
    float mx = fmaxf(fmaxf(fmaxf(tm[0], tm[1]), fmaxf(tm[2], tm[3])),
                     fmaxf(fmaxf(tm[4], tm[5]), fmaxf(tm[6], tm[7])));
    mx = fmaxf(mx, __shfl_xor(mx, 16, 32));
    const float mn    = fmaxf(m, mx);
    const float alpha = fast_exp2(m - mn);
    m = mn;

    // ---- p = exp2(s - mn), spill P^T as f16 (b128/row), tree row-sum
    _Float16* pW = &ldsP[wave * 16 * LDSS];
#pragma unroll
    for (int j = 0; j < 4; ++j) {
#pragma unroll
      for (int r = 0; r < 8; ++r) s[j][r] = fast_exp2(s[j][r] - mn);
      v8h ph;
#pragma unroll
      for (int r = 0; r < 8; ++r) ph[r] = (_Float16)s[j][r];
      *(v8h*)&pW[ln16 * LDSS + j * 16 + 8 * hi] = ph;   // keys contiguous
    }
    v8f a01, a23, as;
#pragma unroll
    for (int r = 0; r < 8; ++r) {
      a01[r] = s[0][r] + s[1][r];
      a23[r] = s[2][r] + s[3][r];
      as[r]  = a01[r] + a23[r];
    }
    float su = ((as[0] + as[1]) + (as[2] + as[3])) +
               ((as[4] + as[5]) + (as[6] + as[7]));
    su += __shfl_xor(su, 16, 32);
    l = l * alpha + su;
#pragma unroll
    for (int j = 0; j < 4; ++j)
#pragma unroll
      for (int r = 0; r < 8; ++r) o[j][r] *= alpha;

    // ---- O^T += V^T x P^T : preload all 4 V^T frags per k-step, then wmma
#pragma unroll
    for (int kk = 0; kk < 2; ++kk) {
      const v16h pb  = load_frag16(&pW[ln16 * LDSS + kk * 32 + hi * 8]);
      const v16h va0 = load_frag16(&ldsV[(0 * 16 + ln16) * LDSS + kk * 32 + hi * 8]);
      const v16h va1 = load_frag16(&ldsV[(1 * 16 + ln16) * LDSS + kk * 32 + hi * 8]);
      const v16h va2 = load_frag16(&ldsV[(2 * 16 + ln16) * LDSS + kk * 32 + hi * 8]);
      const v16h va3 = load_frag16(&ldsV[(3 * 16 + ln16) * LDSS + kk * 32 + hi * 8]);
      o[0] = wmma_f16(va0, pb, o[0]);
      o[1] = wmma_f16(va1, pb, o[1]);
      o[2] = wmma_f16(va2, pb, o[2]);
      o[3] = wmma_f16(va3, pb, o[3]);
    }
    __syncthreads();  // all waves done with K/V before restaging
  }

  // ---- epilogue: per lane, 8 contiguous dims per j-tile -> b128 stores
  const float inv = 1.0f / l;
#pragma unroll
  for (int j = 0; j < 4; ++j) {
    float4 f0 = {o[j][0] * inv, o[j][1] * inv, o[j][2] * inv, o[j][3] * inv};
    float4 f1 = {o[j][4] * inv, o[j][5] * inv, o[j][6] * inv, o[j][7] * inv};
    float* op = Ob + (size_t)qg * D_HEAD + j * 16 + 8 * hi;
    *(float4*)(op)     = f0;
    *(float4*)(op + 4) = f1;
  }
}

extern "C" void kernel_launch(void* const* d_in, const int* in_sizes, int n_in,
                              void* d_out, int out_size, void* d_ws, size_t ws_size,
                              hipStream_t stream) {
  const float* Q = (const float*)d_in[0];
  const float* K = (const float*)d_in[1];
  const float* V = (const float*)d_in[2];
  float*       O = (float*)d_out;
  dim3 grid(S_SEQ / QBLK, N_HEADS, N_BATCH);  // 16 x 16 x 2 = 512 workgroups
  fa_causal_gfx1250<<<grid, 256, 0, stream>>>(Q, K, V, O);
}
